// ContextEncoder_47622597378171
// MI455X (gfx1250) — compile-verified, matrix operand order checked by reading, not run on hardware
//
#include <hip/hip_runtime.h>

#define FDIM 134
#define FPAD 160           // K padded to 5 x 32
#define DHID 128
#define DEMB 128
#define NPOOL 4
#define OUTC (NPOOL * DEMB + 7)   // 519
#define EPSF 1e-16f

typedef __bf16 bf16_t;
typedef bf16_t v8bf  __attribute__((ext_vector_type(8)));
typedef bf16_t v16bf __attribute__((ext_vector_type(16)));
typedef float  v8f   __attribute__((ext_vector_type(8)));

static __device__ __forceinline__ bf16_t f2bf(float f) {
    union { float f; unsigned u; } x; x.f = f;
    unsigned r = x.u + 0x7FFFu + ((x.u >> 16) & 1u);   // round-to-nearest-even
    union { unsigned short s; bf16_t b; } y;
    y.s = (unsigned short)(r >> 16);
    return y.b;
}

// Fragment load for 16x32 bf16 A (or symmetric 32x16 B from transposed weights):
// lane half g holds K = kbase+g*8..+7 in VGPR0-3 and K = kbase+16+g*8..+7 in VGPR4-7.
static __device__ __forceinline__ v16bf load_frag(const bf16_t* __restrict__ p,
                                                  int stride, int row, int kbase, int g) {
    const bf16_t* q = p + row * stride + kbase + g * 8;
    v8bf lo = *reinterpret_cast<const v8bf*>(q);
    v8bf hi = *reinterpret_cast<const v8bf*>(q + 16);
    return __builtin_shufflevector(lo, hi, 0,1,2,3,4,5,6,7,8,9,10,11,12,13,14,15);
}

static __device__ __forceinline__ v8f wmma_bf16(v16bf a, v16bf b, v8f c) {
    return __builtin_amdgcn_wmma_f32_16x16x32_bf16(false, a, false, b, (short)0, c,
                                                   false, false);
}

// ---------------- weight prep: bf16 convert + transpose + K-pad ----------------
__global__ void prep_kernel(const float* __restrict__ gw1, const float* __restrict__ nw1,
                            const float* __restrict__ nw2,
                            bf16_t* __restrict__ w1g, bf16_t* __restrict__ w1n,
                            bf16_t* __restrict__ w2n) {
    int i = blockIdx.x * blockDim.x + threadIdx.x;
    const int n1 = NPOOL * DHID * FPAD;           // 81920
    if (i < n1) {
        int f = i % FPAD;
        int t = i / FPAD;
        int h = t % DHID;
        int k = t / DHID;
        float a = 0.f, b = 0.f;
        if (f < FDIM) {
            a = gw1[(k * FDIM + f) * DHID + h];
            b = nw1[(k * FDIM + f) * DHID + h];
        }
        w1g[i] = f2bf(a);
        w1n[i] = f2bf(b);
    }
    const int n2 = NPOOL * DEMB * DHID;           // 65536
    if (i < n2) {
        int h = i % DHID;
        int t = i / DHID;
        int d = t % DEMB;
        int k = t / DEMB;
        w2n[i] = f2bf(nw2[(k * DHID + h) * DEMB + d]);
    }
}

// ---------------- pass 1: gate scores (x @ gate_w1, relu, . gate_w2) ----------------
__global__ __launch_bounds__(256) void scores_kernel(
        const float* __restrict__ x, const bf16_t* __restrict__ w1t,
        const float* __restrict__ b1, const float* __restrict__ w2,
        float* __restrict__ scores, int nTot) {
    __shared__ __align__(16) bf16_t xlds[64 * FPAD];
    const int tid  = threadIdx.x;
    const int base = blockIdx.x * 64;

    for (int i = tid; i < 64 * FPAD; i += 256) {
        int r = i / FPAD, c = i - r * FPAD;
        int node = base + r;
        float v = 0.f;
        if (node < nTot && c < FDIM) v = x[(size_t)node * FDIM + c];
        xlds[i] = f2bf(v);
    }
    __syncthreads();

    const int lane = tid & 31;
    const int w    = tid >> 5;       // wave id = N-tile
    const int g    = lane >> 4;
    const int ln   = lane & 15;
    const int hcol = w * 16 + ln;

    for (int k = 0; k < NPOOL; ++k) {
        v8f acc[4] = {};
        const bf16_t* wt = w1t + (size_t)k * DHID * FPAD;
        for (int ks = 0; ks < FPAD / 32; ++ks) {
            v16bf bf = load_frag(wt, FPAD, hcol, ks * 32, g);
#pragma unroll
            for (int mt = 0; mt < 4; ++mt) {
                v16bf af = load_frag(xlds, FPAD, mt * 16 + ln, ks * 32, g);
                acc[mt] = wmma_bf16(af, bf, acc[mt]);
            }
        }
        float b1v = b1[k * DHID + hcol];
        float w2v = w2[k * DHID + hcol];
#pragma unroll
        for (int mt = 0; mt < 4; ++mt) {
#pragma unroll
            for (int j = 0; j < 8; ++j) {
                float p = fmaxf(acc[mt][j] + b1v, 0.f) * w2v;
                p += __shfl_xor(p, 1, 32);
                p += __shfl_xor(p, 2, 32);
                p += __shfl_xor(p, 4, 32);
                p += __shfl_xor(p, 8, 32);
                if (ln == 0) {
                    int node = base + mt * 16 + j + 8 * g;
                    if (node < nTot) atomicAdd(scores + (size_t)node * NPOOL + k, p);
                }
            }
        }
    }
}

// ---------------- pass 2: per-graph segment max / sum-exp + output tail ----------------
__global__ __launch_bounds__(256) void stats_kernel(
        const float* __restrict__ scores, const int* __restrict__ batch, int nTot,
        float* __restrict__ mOut, float* __restrict__ invden, float* __restrict__ out,
        const float* __restrict__ t0, const float* __restrict__ t1,
        const float* __restrict__ t2, const float* __restrict__ t3,
        const float* __restrict__ t4, const float* __restrict__ t5,
        const float* __restrict__ t6) {
    __shared__ float red[256];
    const int b   = blockIdx.x;
    const int tid = threadIdx.x;

    int lo = 0, hi = nTot;
    while (lo < hi) { int mid = (lo + hi) >> 1; if (batch[mid] < b) lo = mid + 1; else hi = mid; }
    const int start = lo;
    hi = nTot;
    while (lo < hi) { int mid = (lo + hi) >> 1; if (batch[mid] < b + 1) lo = mid + 1; else hi = mid; }
    const int end = lo;

    const int k   = tid & 3;
    const int sub = tid >> 2;
    float mx = -3.4e38f;
    for (int i = start + sub; i < end; i += 64)
        mx = fmaxf(mx, scores[(size_t)i * NPOOL + k]);
    red[tid] = mx;
    __syncthreads();
    for (int s = 128; s >= 4; s >>= 1) {
        if (tid < s) red[tid] = fmaxf(red[tid], red[tid + s]);
        __syncthreads();
    }
    const float mk = red[k];
    __syncthreads();

    float sm = 0.f;
    for (int i = start + sub; i < end; i += 64)
        sm += expf(scores[(size_t)i * NPOOL + k] - mk);
    red[tid] = sm;
    __syncthreads();
    for (int s = 128; s >= 4; s >>= 1) {
        if (tid < s) red[tid] += red[tid + s];
        __syncthreads();
    }
    if (tid < 4) {
        mOut[b * 4 + tid]   = mk;
        invden[b * 4 + tid] = 1.0f / (red[tid] + EPSF);
    }
    if (tid == 0) {
        float* t = out + (size_t)b * OUTC + NPOOL * DEMB;
        t[0] = t0[b]; t[1] = t1[b]; t[2] = t2[b]; t[3] = t3[b];
        t[4] = t4[b]; t[5] = t5[b]; t[6] = t6[b];
    }
}

// ---------------- pass 3: h1 = relu(x@nn_w1+b1); v = h1@nn_w2+b2; gated segment sum ----
__global__ __launch_bounds__(256) void pool_kernel(
        const float* __restrict__ x, const int* __restrict__ batch,
        const float* __restrict__ scores, const float* __restrict__ mIn,
        const float* __restrict__ invden,
        const bf16_t* __restrict__ w1t, const float* __restrict__ b1,
        const bf16_t* __restrict__ w2t, const float* __restrict__ b2,
        float* __restrict__ out, int nTot) {
    __shared__ __align__(16) bf16_t xlds[64 * FPAD];
    __shared__ __align__(16) bf16_t h1lds[64 * DHID];
    __shared__ float gateL[64 * NPOOL];
    __shared__ int   batchL[64];
    const int tid  = threadIdx.x;
    const int base = blockIdx.x * 64;

    for (int i = tid; i < 64 * FPAD; i += 256) {
        int r = i / FPAD, c = i - r * FPAD;
        int node = base + r;
        float v = 0.f;
        if (node < nTot && c < FDIM) v = x[(size_t)node * FDIM + c];
        xlds[i] = f2bf(v);
    }
    {
        int nl = tid >> 2, kk = tid & 3;
        int node = base + nl;
        float gv = 0.f;
        int bb;
        if (node < nTot) {
            bb = batch[node];
            gv = expf(scores[(size_t)node * NPOOL + kk] - mIn[bb * NPOOL + kk]) *
                 invden[bb * NPOOL + kk];
        } else {
            bb = batch[nTot - 1];
        }
        gateL[nl * NPOOL + kk] = gv;
        if (kk == 0) batchL[nl] = bb;
    }
    __syncthreads();

    const int lane = tid & 31;
    const int w    = tid >> 5;
    const int g    = lane >> 4;
    const int ln   = lane & 15;
    const bool same = (batchL[0] == batchL[63]);   // block-uniform
    const int  bSame = batchL[0];

    for (int k = 0; k < NPOOL; ++k) {
        // --- phase 1: h1 tile (this wave owns N-tile w) ---
        {
            const int hcol = w * 16 + ln;
            v8f acc[4] = {};
            const bf16_t* wt = w1t + (size_t)k * DHID * FPAD;
            for (int ks = 0; ks < FPAD / 32; ++ks) {
                v16bf bf = load_frag(wt, FPAD, hcol, ks * 32, g);
#pragma unroll
                for (int mt = 0; mt < 4; ++mt) {
                    v16bf af = load_frag(xlds, FPAD, mt * 16 + ln, ks * 32, g);
                    acc[mt] = wmma_bf16(af, bf, acc[mt]);
                }
            }
            float b1v = b1[k * DHID + hcol];
#pragma unroll
            for (int mt = 0; mt < 4; ++mt)
#pragma unroll
                for (int j = 0; j < 8; ++j) {
                    int row = mt * 16 + j + 8 * g;
                    h1lds[row * DHID + hcol] = f2bf(fmaxf(acc[mt][j] + b1v, 0.f));
                }
        }
        __syncthreads();
        // --- phase 2: v tile, gate-scale, segment accumulate ---
        {
            const int dcol = w * 16 + ln;
            v8f acc[4] = {};
            const bf16_t* wt = w2t + (size_t)k * DEMB * DHID;
            for (int ks = 0; ks < DHID / 32; ++ks) {
                v16bf bf = load_frag(wt, DHID, dcol, ks * 32, g);
#pragma unroll
                for (int mt = 0; mt < 4; ++mt) {
                    v16bf af = load_frag(h1lds, DHID, mt * 16 + ln, ks * 32, g);
                    acc[mt] = wmma_bf16(af, bf, acc[mt]);
                }
            }
            float b2v = b2[k * DEMB + dcol];
            if (same) {
                float colsum = 0.f;
#pragma unroll
                for (int mt = 0; mt < 4; ++mt)
#pragma unroll
                    for (int j = 0; j < 8; ++j) {
                        int row = mt * 16 + j + 8 * g;
                        colsum += gateL[row * NPOOL + k] * (acc[mt][j] + b2v);
                    }
                atomicAdd(out + (size_t)bSame * OUTC + k * DEMB + dcol, colsum);
            } else {
#pragma unroll
                for (int mt = 0; mt < 4; ++mt)
#pragma unroll
                    for (int j = 0; j < 8; ++j) {
                        int row = mt * 16 + j + 8 * g;
                        float c = gateL[row * NPOOL + k] * (acc[mt][j] + b2v);
                        atomicAdd(out + (size_t)batchL[row] * OUTC + k * DEMB + dcol, c);
                    }
            }
        }
        __syncthreads();
    }
}

extern "C" void kernel_launch(void* const* d_in, const int* in_sizes, int n_in,
                              void* d_out, int out_size, void* d_ws, size_t ws_size,
                              hipStream_t stream) {
    const float* x    = (const float*)d_in[0];
    const int*   bat  = (const int*)d_in[1];
    const float* nn   = (const float*)d_in[2];
    const float* Om   = (const float*)d_in[3];
    const float* Ph   = (const float*)d_in[4];
    const float* La   = (const float*)d_in[5];
    const float* Omn  = (const float*)d_in[6];
    const float* Phn  = (const float*)d_in[7];
    const float* Lan  = (const float*)d_in[8];
    const float* gw1  = (const float*)d_in[9];
    const float* gb1  = (const float*)d_in[10];
    const float* gw2  = (const float*)d_in[11];
    // d_in[12] = gate_b2: constant per-head shift, cancels in segment softmax
    const float* nw1  = (const float*)d_in[13];
    const float* nb1  = (const float*)d_in[14];
    const float* nw2  = (const float*)d_in[15];
    const float* nb2  = (const float*)d_in[16];

    const int nTot = in_sizes[1];     // 250000 nodes
    const int B    = in_sizes[2];     // 256 graphs
    float* out = (float*)d_out;

    char* ws = (char*)d_ws;
    float* scores = (float*)ws;  ws += (size_t)nTot * 4 * sizeof(float);
    float* mBuf   = (float*)ws;  ws += (size_t)B * 4 * sizeof(float);
    float* invden = (float*)ws;  ws += (size_t)B * 4 * sizeof(float);
    bf16_t* w1g   = (bf16_t*)ws; ws += (size_t)NPOOL * DHID * FPAD * sizeof(bf16_t);
    bf16_t* w1n   = (bf16_t*)ws; ws += (size_t)NPOOL * DHID * FPAD * sizeof(bf16_t);
    bf16_t* w2n   = (bf16_t*)ws;

    hipMemsetAsync(scores, 0, (size_t)nTot * 4 * sizeof(float), stream);
    hipMemsetAsync(d_out, 0, (size_t)out_size * sizeof(float), stream);

    prep_kernel<<<(NPOOL * DHID * FPAD + 255) / 256, 256, 0, stream>>>(
        gw1, nw1, nw2, w1g, w1n, w2n);

    const int nb = (nTot + 63) / 64;
    scores_kernel<<<nb, 256, 0, stream>>>(x, w1g, gb1, gw2, scores, nTot);
    stats_kernel<<<B, 256, 0, stream>>>(scores, bat, nTot, mBuf, invden, out,
                                        nn, Om, Ph, La, Omn, Phn, Lan);
    pool_kernel<<<nb, 256, 0, stream>>>(x, bat, scores, mBuf, invden,
                                        w1n, nb1, w2n, nb2, out, nTot);
}
